// TinyAttention_10849087390332
// MI455X (gfx1250) — compile-verified, matrix operand order checked by reading, not run on hardware
//
#include <hip/hip_runtime.h>
#include <hip/hip_bf16.h>
#include <hip/hip_fp16.h>

typedef __attribute__((ext_vector_type(16))) _Float16 v16h;
typedef __attribute__((ext_vector_type(8)))  _Float16 v8h;
typedef __attribute__((ext_vector_type(8)))  float    v8f;
typedef __attribute__((ext_vector_type(4)))  unsigned int u32x4;
typedef __attribute__((ext_vector_type(8)))  int      i32x8;
typedef __attribute__((ext_vector_type(4)))  int      i32x4;

#define B_   2
#define L_   2048
#define D_   1024
#define H_   16
#define HD_  64
#define ND3  (3 * D_)

// ---------------------------------------------------------------------------
// Unified fragment loader (CDNA5 ISA 7.12.2, 16-bit operands).
// A lane's 16 halves are two contiguous 8-half runs of one row:
//   k = kbase + 8*half + {0..7}  and  k + 16.
// All tiles are staged so the fragment's "row" is contiguous in LDS
// (B operands staged transposed), giving 2x ds_load_b128 per fragment.
// ld (in halves) must be a multiple of 8 so rows stay 16B-aligned.
// ---------------------------------------------------------------------------
__device__ __forceinline__ v16h frag16(const _Float16* base, int ld, int kbase) {
  const int lane = threadIdx.x & 31;
  const int half = lane >> 4;
  const int row  = lane & 15;
  const _Float16* p = base + row * ld + kbase + 8 * half;
  const v8h a = *(const v8h*)p;
  const v8h b = *(const v8h*)(p + 16);
  return __builtin_shufflevector(a, b, 0, 1, 2, 3, 4, 5, 6, 7,
                                 8, 9, 10, 11, 12, 13, 14, 15);
}

// ---------------------------------------------------------------------------
// Tensor Data Mover: 2-D f16 tile (tile_d0 x tile_d1, row stride in elems)
// from global to LDS, with LDS padding 4 DWORDs per 32 DWORDs (64-half rows
// land on a 72-half LDS stride). D# layout per CDNA5 ISA ch.8.
// ---------------------------------------------------------------------------
__device__ __forceinline__ void tdm_load_2d_f16(unsigned lds_off, const void* gptr,
                                                int tile_d0, int tile_d1,
                                                long long row_stride) {
  const unsigned long long ga = (unsigned long long)(size_t)gptr;
  u32x4 g0;
  g0[0] = 1u;                                         // count=1, user descriptor
  g0[1] = lds_off;                                    // lds_addr (bytes)
  g0[2] = (unsigned)(ga & 0xffffffffu);               // global_addr[95:64]
  g0[3] = (unsigned)((ga >> 32) & 0x1ffffffu)         // global_addr[120:96]
        | (2u << 30);                                 // type = 2 (image)
  i32x8 g1;
  g1[0] = (1 << 16)                                   // data_size = 2 bytes
        | (1 << 20)                                   // pad_enable
        | (4 << 22)                                   // pad_interval: 32 DWORDs
        | (3 << 25);                                  // pad_amount: 4 DWORDs
  g1[1] = (tile_d0 & 0xffff) << 16;                   // tensor_dim0[15:0]
  g1[2] = ((tile_d0 >> 16) & 0xffff)                  // tensor_dim0[31:16]
        | ((tile_d1 & 0xffff) << 16);                 // tensor_dim1[15:0]
  g1[3] = ((tile_d1 >> 16) & 0xffff)                  // tensor_dim1[31:16]
        | ((tile_d0 & 0xffff) << 16);                 // tile_dim0
  g1[4] = (tile_d1 & 0xffff);                         // tile_dim1 (tile_dim2=0)
  g1[5] = (int)(row_stride & 0xffffffffLL);           // tensor_dim0_stride lo
  g1[6] = (int)((row_stride >> 32) & 0xffffLL);       // tensor_dim0_stride hi
  g1[7] = 0;                                          // tensor_dim1_stride (2-D)
  i32x4 z4;
  z4[0] = z4[1] = z4[2] = z4[3] = 0;
#if defined(__clang_major__) && (__clang_major__ >= 23)
  i32x8 z8;
#pragma unroll
  for (int i = 0; i < 8; ++i) z8[i] = 0;
  __builtin_amdgcn_tensor_load_to_lds(g0, g1, z4, z4, z8, 0);
#else
  __builtin_amdgcn_tensor_load_to_lds(g0, g1, z4, z4, 0);
#endif
}

// ---------------------------------------------------------------------------
// Kernel 1: QKV projection  (M=B*L=4096, K=1024, N=3072), fp32 in -> f16 out
// 128x64 tile/block, 8 waves; A staged row-major, B staged transposed [n][k].
// ---------------------------------------------------------------------------
__global__ __launch_bounds__(256)
void qkv_gemm_kernel(const float* __restrict__ x, const float* __restrict__ w,
                     _Float16* __restrict__ qkv) {
  __shared__ __align__(16) _Float16 As[128 * 40];   // [m][k]
  __shared__ __align__(16) _Float16 Bs[64 * 40];    // [n][k] (transposed)
  const int tid  = threadIdx.x;
  const int wave = tid >> 5;
  const int lane = tid & 31;
  const int half = lane >> 4;
  const int row0 = blockIdx.x * 128;
  const int col0 = blockIdx.y * 64;

  v8f acc[4];
#pragma unroll
  for (int t = 0; t < 4; ++t)
#pragma unroll
    for (int f = 0; f < 8; ++f) acc[t][f] = 0.0f;

  for (int k0 = 0; k0 < D_; k0 += 32) {
    if (k0 + 32 < D_) {                      // prefetch next tiles
      __builtin_prefetch(&x[(size_t)(row0 + (tid >> 1)) * D_ + k0 + 32], 0, 1);
      __builtin_prefetch(&w[(size_t)(k0 + 32 + (tid >> 3)) * ND3 + col0 + (tid & 7) * 8], 0, 1);
    }
#pragma unroll
    for (int i = 0; i < 16; ++i) {           // A: 128x32
      const int idx = i * 256 + tid;
      const int r = idx >> 5, c = idx & 31;
      As[r * 40 + c] = (_Float16)x[(size_t)(row0 + r) * D_ + k0 + c];
    }
#pragma unroll
    for (int i = 0; i < 8; ++i) {            // B: 32x64, stored [n][k]
      const int idx = i * 256 + tid;
      const int r = idx >> 6, c = idx & 63;
      Bs[c * 40 + r] = (_Float16)w[(size_t)(k0 + r) * ND3 + col0 + c];
    }
    __syncthreads();

    const v16h a = frag16(As + (wave * 16) * 40, 40, 0);
#pragma unroll
    for (int t = 0; t < 4; ++t) {
      const v16h b = frag16(Bs + (t * 16) * 40, 40, 0);
      acc[t] = __builtin_amdgcn_wmma_f32_16x16x32_f16(
          false, a, false, b, (short)0, acc[t], false, false);
    }
    __syncthreads();
  }

#pragma unroll
  for (int t = 0; t < 4; ++t)
#pragma unroll
    for (int f = 0; f < 8; ++f) {
      const int m = row0 + wave * 16 + f + 8 * half;
      const int n = col0 + t * 16 + (lane & 15);
      qkv[(size_t)m * ND3 + n] = (_Float16)acc[t][f];
    }
}

// ---------------------------------------------------------------------------
// Kernel 2: RoPE on q and k (pairs j, j+32), in place on the f16 qkv buffer.
// ---------------------------------------------------------------------------
__global__ __launch_bounds__(256)
void rope_kernel(_Float16* __restrict__ qkv) {
  const int idx = blockIdx.x * blockDim.x + threadIdx.x;
  if (idx >= B_ * L_ * H_ * 32) return;
  const int j = idx & 31;
  const int h = (idx >> 5) & (H_ - 1);
  const int l = (idx >> 9) & (L_ - 1);
  const int b = idx >> 20;

  const float inv_freq = __powf(10000.0f, -(float)(2 * j) / (float)HD_);
  const float ang = (float)l * inv_freq;
  const float c = __cosf(ang);
  const float s = __sinf(ang);

  const size_t base = ((size_t)(b * L_ + l)) * ND3 + h * HD_;
#pragma unroll
  for (int sidx = 0; sidx < 2; ++sidx) {     // 0 = q, 1 = k
    _Float16* p = qkv + base + sidx * D_;
    const float x1 = (float)p[j];
    const float x2 = (float)p[j + 32];
    p[j]      = (_Float16)(x1 * c - x2 * s);
    p[j + 32] = (_Float16)(x2 * c + x1 * s);
  }
}

// ---------------------------------------------------------------------------
// Kernel 3: flash attention. One block per (b, h, 64-query tile), 4 waves.
// Q/K tiles arrive via TENSOR_LOAD_TO_LDS (72-half padded stride); V is
// staged transposed [d][key] so P·V B-fragments are contiguous; P is written
// row-major [q][key] so it reloads as an A operand with b128 loads.
// ---------------------------------------------------------------------------
__global__ __launch_bounds__(128)
void attn_kernel(const _Float16* __restrict__ qkv, _Float16* __restrict__ o) {
  __shared__ __align__(16) _Float16 Qs[64 * 72];   // [q][d]
  __shared__ __align__(16) _Float16 Ks[64 * 72];   // [key][d]  (acts as B^T)
  __shared__ __align__(16) _Float16 Vt[64 * 72];   // [d][key]  (transposed)
  __shared__ __align__(16) _Float16 Ps[64 * 72];   // [q][key]

  const int tid   = threadIdx.x;
  const int wave  = tid >> 5;
  const int lane  = tid & 31;
  const int half  = lane >> 4;
  const int nlane = lane & 15;

  const int blk = blockIdx.x;                // b * H * (L/64)
  const int qt  = blk & 31;                  // L/64 = 32
  const int h   = (blk >> 5) & (H_ - 1);
  const int b   = blk >> 9;
  const int q0  = qt * 64;

  const _Float16* qbase = qkv + (size_t)b * L_ * ND3 + h * HD_;
  const _Float16* kbase = qbase + D_;
  const _Float16* vbase = qbase + 2 * D_;

  if (wave == 0) {                           // Q tile via TDM
    tdm_load_2d_f16((unsigned)(size_t)Qs, qbase + (size_t)q0 * ND3, 64, 64, ND3);
    __builtin_amdgcn_s_wait_tensorcnt(0);
  }
  __syncthreads();

  float m_i[8], l_i[8];
#pragma unroll
  for (int f = 0; f < 8; ++f) { m_i[f] = -3.0e38f; l_i[f] = 0.0f; }
  v8f acc_o[4];
#pragma unroll
  for (int t = 0; t < 4; ++t)
#pragma unroll
    for (int f = 0; f < 8; ++f) acc_o[t][f] = 0.0f;

  const float scale = 0.125f;                // 1/sqrt(64)

  for (int kt = 0; kt < L_ / 64; ++kt) {
    const int kk0 = kt * 64;
    __syncthreads();                         // previous iteration's reads done
    if (wave == 0)                           // K tile via TDM (async w/ V stage)
      tdm_load_2d_f16((unsigned)(size_t)Ks, kbase + (size_t)kk0 * ND3, 64, 64, ND3);
#pragma unroll
    for (int i = 0; i < 32; ++i) {           // V tile staged transposed
      const int idx = i * 128 + tid;
      const int r = idx >> 6, c = idx & 63;  // r = key, c = d
      Vt[c * 72 + r] = vbase[(size_t)(kk0 + r) * ND3 + c];
    }
    if (wave == 0) __builtin_amdgcn_s_wait_tensorcnt(0);
    __syncthreads();

    // S = Q K^T -> 16x64 per wave; K rows are B^T fragment rows
    v8f acc_s[4];
#pragma unroll
    for (int t = 0; t < 4; ++t)
#pragma unroll
      for (int f = 0; f < 8; ++f) acc_s[t][f] = 0.0f;
#pragma unroll
    for (int ks = 0; ks < 2; ++ks) {
      const v16h a = frag16(Qs + (wave * 16) * 72, 72, ks * 32);
#pragma unroll
      for (int t = 0; t < 4; ++t) {
        const v16h bfr = frag16(Ks + (t * 16) * 72, 72, ks * 32);
        acc_s[t] = __builtin_amdgcn_wmma_f32_16x16x32_f16(
            false, a, false, bfr, (short)0, acc_s[t], false, false);
      }
    }

    // Online softmax per row (row M = f + 8*half)
#pragma unroll
    for (int f = 0; f < 8; ++f) {
      float vmax = -3.0e38f;
#pragma unroll
      for (int t = 0; t < 4; ++t) vmax = fmaxf(vmax, acc_s[t][f]);
#pragma unroll
      for (int mask = 1; mask < 16; mask <<= 1)
        vmax = fmaxf(vmax, __shfl_xor(vmax, mask, 32));
      vmax *= scale;
      const float mnew  = fmaxf(m_i[f], vmax);
      const float alpha = __expf(m_i[f] - mnew);
      float rsum = 0.0f;
#pragma unroll
      for (int t = 0; t < 4; ++t) {
        const float p = __expf(acc_s[t][f] * scale - mnew);
        rsum += p;
        Ps[(wave * 16 + f + 8 * half) * 72 + t * 16 + nlane] = (_Float16)p;
      }
#pragma unroll
      for (int mask = 1; mask < 16; mask <<= 1)
        rsum += __shfl_xor(rsum, mask, 32);
      l_i[f] = l_i[f] * alpha + rsum;
      m_i[f] = mnew;
#pragma unroll
      for (int t = 0; t < 4; ++t) acc_o[t][f] *= alpha;
    }

    // O += P V : A = P rows (wave-local), B = Vt rows (d-major)
#pragma unroll
    for (int ks = 0; ks < 2; ++ks) {
      const v16h a = frag16(Ps + (wave * 16) * 72, 72, ks * 32);
#pragma unroll
      for (int t = 0; t < 4; ++t) {
        const v16h bfr = frag16(Vt + (t * 16) * 72, 72, ks * 32);
        acc_o[t] = __builtin_amdgcn_wmma_f32_16x16x32_f16(
            false, a, false, bfr, (short)0, acc_o[t], false, false);
      }
    }
  }

  // Normalize and store o as f16 [B*L][D], head-interleaved
#pragma unroll
  for (int t = 0; t < 4; ++t)
#pragma unroll
    for (int f = 0; f < 8; ++f) {
      const int row = q0 + wave * 16 + f + 8 * half;
      const int d   = t * 16 + nlane;
      o[((size_t)b * L_ + row) * D_ + h * HD_ + d] =
          (_Float16)(acc_o[t][f] / l_i[f]);
    }
}

// ---------------------------------------------------------------------------
// Kernel 4: output projection (M=4096, K=1024, N=1024), f16 A, fp32 out
// ---------------------------------------------------------------------------
__global__ __launch_bounds__(256)
void out_gemm_kernel(const _Float16* __restrict__ o, const float* __restrict__ w,
                     float* __restrict__ out) {
  __shared__ __align__(16) _Float16 As[128 * 40];   // [m][k]
  __shared__ __align__(16) _Float16 Bs[64 * 40];    // [n][k] (transposed)
  const int tid  = threadIdx.x;
  const int wave = tid >> 5;
  const int lane = tid & 31;
  const int half = lane >> 4;
  const int row0 = blockIdx.x * 128;
  const int col0 = blockIdx.y * 64;

  v8f acc[4];
#pragma unroll
  for (int t = 0; t < 4; ++t)
#pragma unroll
    for (int f = 0; f < 8; ++f) acc[t][f] = 0.0f;

  for (int k0 = 0; k0 < D_; k0 += 32) {
    if (k0 + 32 < D_) {
      __builtin_prefetch(&o[(size_t)(row0 + (tid >> 1)) * D_ + k0 + 32], 0, 1);
      __builtin_prefetch(&w[(size_t)(k0 + 32 + (tid >> 3)) * D_ + col0 + (tid & 7) * 8], 0, 1);
    }
#pragma unroll
    for (int i = 0; i < 16; ++i) {
      const int idx = i * 256 + tid;
      const int r = idx >> 5, c = idx & 31;
      As[r * 40 + c] = o[(size_t)(row0 + r) * D_ + k0 + c];
    }
#pragma unroll
    for (int i = 0; i < 8; ++i) {
      const int idx = i * 256 + tid;
      const int r = idx >> 6, c = idx & 63;
      Bs[c * 40 + r] = (_Float16)w[(size_t)(k0 + r) * D_ + col0 + c];
    }
    __syncthreads();

    const v16h a = frag16(As + (wave * 16) * 40, 40, 0);
#pragma unroll
    for (int t = 0; t < 4; ++t) {
      const v16h b = frag16(Bs + (t * 16) * 40, 40, 0);
      acc[t] = __builtin_amdgcn_wmma_f32_16x16x32_f16(
          false, a, false, b, (short)0, acc[t], false, false);
    }
    __syncthreads();
  }

#pragma unroll
  for (int t = 0; t < 4; ++t)
#pragma unroll
    for (int f = 0; f < 8; ++f) {
      const int m = row0 + wave * 16 + f + 8 * half;
      const int n = col0 + t * 16 + (lane & 15);
      out[(size_t)m * D_ + n] = acc[t][f];
    }
}

// ---------------------------------------------------------------------------
extern "C" void kernel_launch(void* const* d_in, const int* in_sizes, int n_in,
                              void* d_out, int out_size, void* d_ws, size_t ws_size,
                              hipStream_t stream) {
  const float* x     = (const float*)d_in[0];
  const float* w_qkv = (const float*)d_in[1];
  const float* w_out = (const float*)d_in[2];
  float* out = (float*)d_out;

  _Float16* qkv = (_Float16*)d_ws;                       // B*L*3D f16 = 24 MB
  _Float16* o   = qkv + (size_t)B_ * L_ * ND3;           // B*L*D  f16 =  8 MB

  qkv_gemm_kernel<<<dim3((B_ * L_) / 128, ND3 / 64), 256, 0, stream>>>(x, w_qkv, qkv);
  rope_kernel<<<(B_ * L_ * H_ * 32) / 256, 256, 0, stream>>>(qkv);
  attn_kernel<<<B_ * H_ * (L_ / 64), 128, 0, stream>>>(qkv, o);
  out_gemm_kernel<<<dim3((B_ * L_) / 128, D_ / 64), 256, 0, stream>>>(o, w_out, out);
}
